// MultiHeadAttention_16028817949023
// MI455X (gfx1250) — matrix operand order repaired and checked by reading, not compiled, not run on hardware
//
#include <hip/hip_runtime.h>
#include <hip/hip_bf16.h>
#include <math.h>

typedef __attribute__((ext_vector_type(16))) _Float16 v16h;
typedef __attribute__((ext_vector_type(8)))  _Float16 v8h;
typedef __attribute__((ext_vector_type(4)))  _Float16 v4h;
typedef __attribute__((ext_vector_type(8)))  float    v8f;

#define BATCH 4
#define SEQ   2048
#define DIM   1024
#define NH    16
#define DH    64
#define MTOT  (BATCH * SEQ)   // 8192

// GEMM tiling: block 128x128, 8 waves as 4(M) x 2(N), wave tile 32x64.
#define TM 128
#define TN 128
#define TK 32

static __device__ __forceinline__ v8f wmma_f16(v16h a, v16h b, v8f c) {
  return __builtin_amdgcn_wmma_f32_16x16x32_f16(false, a, false, b, (short)0, c,
                                                false, false);
}

// ---------------------------------------------------------------------------
// Kernel 1: xout[b,h,s,dh] = scale * (X[m,:] . W[n,:])   (fp32 in, f16 out)
// ---------------------------------------------------------------------------
__global__ __launch_bounds__(256) void mha_proj_qkv(
    const float* __restrict__ X, const float* __restrict__ W,
    _Float16* __restrict__ out, float scale) {
  __shared__ _Float16 sA[TM * TK];  // [m][k]
  __shared__ _Float16 sB[TN * TK];  // [n][k]

  const int tid   = threadIdx.x;
  const int lane  = tid & 31;
  const int wave  = tid >> 5;      // 0..7
  const int waveM = wave >> 1;     // 0..3  (32 rows each)
  const int waveN = wave & 1;      // 0..1  (64 cols each)
  const int bm = blockIdx.x * TM;
  const int bn = blockIdx.y * TN;

  v8f acc[2][4] = {};

  const int arow = lane & 15;
  const int akb  = (lane >> 4) * 16;   // A kbase
  const int bkb  = (lane >> 4) * 16;   // B kbase

  for (int k0 = 0; k0 < DIM; k0 += TK) {
    // Load A tile 128x32 fp32 -> f16 LDS (4 float4 per thread)
    for (int i = 0; i < 4; ++i) {
      int idx = tid + i * 256;          // float4 index, 8 per row
      int m   = idx >> 3;
      int kk  = (idx & 7) * 4;
      float4 f = *(const float4*)(X + (size_t)(bm + m) * DIM + k0 + kk);
      v4h hv; hv[0] = (_Float16)f.x; hv[1] = (_Float16)f.y;
              hv[2] = (_Float16)f.z; hv[3] = (_Float16)f.w;
      *(v4h*)&sA[m * TK + kk] = hv;
    }
    // Load B tile 128x32 fp32 -> f16 LDS (4 float4 per thread)
    for (int i = 0; i < 4; ++i) {
      int idx = tid + i * 256;
      int n   = idx >> 3;
      int kk  = (idx & 7) * 4;
      float4 f = *(const float4*)(W + (size_t)(bn + n) * DIM + k0 + kk);
      v4h hv; hv[0] = (_Float16)f.x; hv[1] = (_Float16)f.y;
              hv[2] = (_Float16)f.z; hv[3] = (_Float16)f.w;
      *(v4h*)&sB[n * TK + kk] = hv;
    }
    __syncthreads();

    v16h a[2], b[4];
    for (int i = 0; i < 2; ++i) {
      int m = waveM * 32 + i * 16 + arow;
      v8h lo = *(const v8h*)&sA[m * TK + akb];
      v8h hi = *(const v8h*)&sA[m * TK + akb + 8];
      for (int e = 0; e < 8; ++e) { a[i][e] = lo[e]; a[i][8 + e] = hi[e]; }
    }
    for (int j = 0; j < 4; ++j) {
      int n = waveN * 64 + j * 16 + arow;
      b[j] = *(const v16h*)&sB[n * TK + bkb];
    }
    for (int i = 0; i < 2; ++i)
      for (int j = 0; j < 4; ++j)
        acc[i][j] = wmma_f16(a[i], b[j], acc[i][j]);
    __syncthreads();
  }

  // Epilogue: C/D layout row = v + (lane/16)*8, col = lane%16.
  // Write f16 into [B, H, S, DH].
  const int col   = lane & 15;
  const int rbase = (lane >> 4) * 8;
  for (int i = 0; i < 2; ++i)
    for (int j = 0; j < 4; ++j)
      for (int v = 0; v < 8; ++v) {
        int m  = bm + waveM * 32 + i * 16 + rbase + v;
        int n  = bn + waveN * 64 + j * 16 + col;
        int bb = m >> 11;        // / SEQ
        int s  = m & (SEQ - 1);
        int h  = n >> 6;         // / DH
        int dh = n & (DH - 1);
        out[(((size_t)bb * NH + h) * SEQ + s) * DH + dh] =
            (_Float16)(acc[i][j][v] * scale);
      }
}

// ---------------------------------------------------------------------------
// Kernel 2: causal flash attention.  xq already scaled by 1/sqrt(DH).
// Grid: (B*H, SEQ/128). Block 256 = 8 waves, each wave owns 16 query rows.
// Output xo: f16 [B, S, H*DH].
// ---------------------------------------------------------------------------
__global__ __launch_bounds__(256) void mha_flash_attn(
    const _Float16* __restrict__ xq, const _Float16* __restrict__ xk,
    const _Float16* __restrict__ xv, _Float16* __restrict__ xo) {
  __shared__ _Float16 sK[32 * 64];       // [key][d]
  __shared__ _Float16 sVt[64 * 32];      // [d][key]  (transposed)
  __shared__ _Float16 sP[8][16 * 32];    // per-wave P tile [row][key]

  const int bh   = blockIdx.x;
  const int b    = bh >> 4;
  const int h    = bh & 15;
  const int rowBlock = blockIdx.y * 128;
  const int tid  = threadIdx.x;
  const int lane = tid & 31;
  const int wave = tid >> 5;
  const int qrow0 = rowBlock + wave * 16;

  const _Float16* Qb = xq + ((size_t)b * NH + h) * SEQ * DH;
  const _Float16* Kb = xk + ((size_t)b * NH + h) * SEQ * DH;
  const _Float16* Vb = xv + ((size_t)b * NH + h) * SEQ * DH;

  const int arow = lane & 15;
  const int akb  = (lane >> 4) * 16;
  const int bkb  = (lane >> 4) * 16;
  const int colb = lane & 15;
  const int rsel = (lane >> 4) * 8;

  // Load Q fragments once (A layout, straight from global).
  v16h qa[2];
  for (int j = 0; j < 2; ++j) {
    const _Float16* p = Qb + (size_t)(qrow0 + arow) * DH + j * 32 + akb;
    v8h lo = *(const v8h*)p;
    v8h hi = *(const v8h*)(p + 8);
    for (int e = 0; e < 8; ++e) { qa[j][e] = lo[e]; qa[j][8 + e] = hi[e]; }
  }

  v8f  o[4] = {};
  float mrow[8], lrow[8];
  for (int v = 0; v < 8; ++v) { mrow[v] = -1e30f; lrow[v] = 0.0f; }

  const int nkt = (rowBlock + 128) / 32;   // causal bound for this block
  for (int kt = 0; kt < nkt; ++kt) {
    const int key0 = kt * 32;
    // Cooperative load: K tile (32x64) straight; V tile transposed into sVt.
    {
      int kr = tid >> 3;
      int dd = (tid & 7) * 8;
      v8h kv = *(const v8h*)(Kb + (size_t)(key0 + kr) * DH + dd);
      *(v8h*)&sK[kr * 64 + dd] = kv;
      v8h vv = *(const v8h*)(Vb + (size_t)(key0 + kr) * DH + dd);
      for (int e = 0; e < 8; ++e) sVt[(dd + e) * 32 + kr] = vv[e];
    }
    __syncthreads();

    // Fully-masked tiles are a mathematical no-op for the online-softmax
    // update; skip compute (barriers stay uniform across the block).
    if (key0 <= qrow0 + 15) {
      // S = Q @ K^T  (two 16-col tiles, K-dim 64 = 2 WMMAs each)
      v8f sacc[2];
      for (int n0 = 0; n0 < 2; ++n0) {
        v8f z = {};
        for (int j = 0; j < 2; ++j) {
          v16h bf = *(const v16h*)&sK[(n0 * 16 + colb) * 64 + j * 32 + bkb];
          z = wmma_f16(qa[j], bf, z);
        }
        sacc[n0] = z;
      }

      // Causal mask + online softmax (row = qrow0+rsel+v, col striped on lanes)
      float scl[8];
      for (int v = 0; v < 8; ++v) {
        int rowg = qrow0 + rsel + v;
        for (int n0 = 0; n0 < 2; ++n0) {
          int colg = key0 + n0 * 16 + colb;
          float s = sacc[n0][v];
          if (colg > rowg) s = -1e30f;
          sacc[n0][v] = s;
        }
        float m = fmaxf(sacc[0][v], sacc[1][v]);
        for (int off = 1; off < 16; off <<= 1)
          m = fmaxf(m, __shfl_xor(m, off, 32));
        float mnew = fmaxf(mrow[v], m);
        scl[v] = __expf(mrow[v] - mnew);
        mrow[v] = mnew;

        float rs = 0.0f;
        for (int n0 = 0; n0 < 2; ++n0) {
          float p = __expf(sacc[n0][v] - mnew);
          sacc[n0][v] = p;
          rs += p;
        }
        for (int off = 1; off < 16; off <<= 1)
          rs += __shfl_xor(rs, off, 32);
        lrow[v] = lrow[v] * scl[v] + rs;
      }

      // P -> per-wave LDS (C layout -> memory), rescale O accumulators.
      for (int v = 0; v < 8; ++v)
        for (int n0 = 0; n0 < 2; ++n0)
          sP[wave][(rsel + v) * 32 + n0 * 16 + colb] = (_Float16)sacc[n0][v];
      for (int t = 0; t < 4; ++t)
        for (int v = 0; v < 8; ++v)
          o[t][v] *= scl[v];

      __builtin_amdgcn_wave_barrier();   // order P store vs. P reload (same wave)

      // O += P(16x32) @ V(32x64): reload P in A layout; V^T B frags contiguous.
      v16h pa;
      {
        const _Float16* p = &sP[wave][arow * 32 + akb];
        v8h lo = *(const v8h*)p;
        v8h hi = *(const v8h*)(p + 8);
        for (int e = 0; e < 8; ++e) { pa[e] = lo[e]; pa[8 + e] = hi[e]; }
      }
      for (int t = 0; t < 4; ++t) {
        v16h bf = *(const v16h*)&sVt[(t * 16 + colb) * 32 + bkb];
        o[t] = wmma_f16(pa, bf, o[t]);
      }
    }
    __syncthreads();
  }

  // Normalize and store f16 [B, S, H*DH]
  for (int t = 0; t < 4; ++t)
    for (int v = 0; v < 8; ++v) {
      int rowg = qrow0 + rsel + v;
      float val = o[t][v] / lrow[v];
      xo[((size_t)b * SEQ + rowg) * DIM + h * DH + t * 16 + colb] =
          (_Float16)val;
    }
}

// ---------------------------------------------------------------------------
// Kernel 3: out[m,n] = Xh[m,:] . Wo[n,:]   (f16 activations, fp32 weights/out)
// ---------------------------------------------------------------------------
__global__ __launch_bounds__(256) void mha_proj_out(
    const _Float16* __restrict__ Xh, const float* __restrict__ W,
    float* __restrict__ out) {
  __shared__ _Float16 sA[TM * TK];
  __shared__ _Float16 sB[TN * TK];

  const int tid   = threadIdx.x;
  const int lane  = tid & 31;
  const int wave  = tid >> 5;
  const int waveM = wave >> 1;
  const int waveN = wave & 1;
  const int bm = blockIdx.x * TM;
  const int bn = blockIdx.y * TN;

  v8f acc[2][4] = {};
  const int arow = lane & 15;
  const int akb  = (lane >> 4) * 16;
  const int bkb  = (lane >> 4) * 16;

  for (int k0 = 0; k0 < DIM; k0 += TK) {
    // A tile already f16: 4096 halves, 16 per thread = 2 x v8h
    for (int i = 0; i < 2; ++i) {
      int idx = tid + i * 256;        // v8h chunk index, 4 per row
      int m   = idx >> 2;
      int kk  = (idx & 3) * 8;
      *(v8h*)&sA[m * TK + kk] =
          *(const v8h*)(Xh + (size_t)(bm + m) * DIM + k0 + kk);
    }
    // B tile 128x32 fp32 -> f16
    for (int i = 0; i < 4; ++i) {
      int idx = tid + i * 256;
      int n   = idx >> 3;
      int kk  = (idx & 7) * 4;
      float4 f = *(const float4*)(W + (size_t)(bn + n) * DIM + k0 + kk);
      v4h hv; hv[0] = (_Float16)f.x; hv[1] = (_Float16)f.y;
              hv[2] = (_Float16)f.z; hv[3] = (_Float16)f.w;
      *(v4h*)&sB[n * TK + kk] = hv;
    }
    __syncthreads();

    v16h a[2], b[4];
    for (int i = 0; i < 2; ++i) {
      int m = waveM * 32 + i * 16 + arow;
      v8h lo = *(const v8h*)&sA[m * TK + akb];
      v8h hi = *(const v8h*)&sA[m * TK + akb + 8];
      for (int e = 0; e < 8; ++e) { a[i][e] = lo[e]; a[i][8 + e] = hi[e]; }
    }
    for (int j = 0; j < 4; ++j) {
      int n = waveN * 64 + j * 16 + arow;
      b[j] = *(const v16h*)&sB[n * TK + bkb];
    }
    for (int i = 0; i < 2; ++i)
      for (int j = 0; j < 4; ++j)
        acc[i][j] = wmma_f16(a[i], b[j], acc[i][j]);
    __syncthreads();
  }

  const int col   = lane & 15;
  const int rbase = (lane >> 4) * 8;
  for (int i = 0; i < 2; ++i)
    for (int j = 0; j < 4; ++j)
      for (int v = 0; v < 8; ++v) {
        int m = bm + waveM * 32 + i * 16 + rbase + v;
        int n = bn + waveN * 64 + j * 16 + col;
        out[(size_t)m * DIM + n] = acc[i][j][v];
      }
}

// ---------------------------------------------------------------------------
extern "C" void kernel_launch(void* const* d_in, const int* in_sizes, int n_in,
                              void* d_out, int out_size, void* d_ws,
                              size_t ws_size, hipStream_t stream) {
  (void)in_sizes; (void)n_in; (void)out_size; (void)ws_size;

  const float* q  = (const float*)d_in[0];
  const float* k  = (const float*)d_in[1];
  const float* v  = (const float*)d_in[2];
  const float* Wq = (const float*)d_in[3];
  const float* Wk = (const float*)d_in[4];
  const float* Wv = (const float*)d_in[5];
  const float* Wo = (const float*)d_in[6];
  float* out = (float*)d_out;

  const size_t nElem = (size_t)BATCH * NH * SEQ * DH;  // 8M f16 elems
  _Float16* xq = (_Float16*)d_ws;
  _Float16* xk = xq + nElem;
  _Float16* xv = xk + nElem;
  _Float16* xo = xv + nElem;

  dim3 blk(256);
  dim3 gGemm(MTOT / TM, DIM / TN);     // (64, 8)
  dim3 gAttn(BATCH * NH, SEQ / 128);   // (64, 16)

  const float qscale = 1.0f / 8.0f;    // 1/sqrt(DH)
  mha_proj_qkv<<<gGemm, blk, 0, stream>>>(q, Wq, xq, qscale);
  mha_proj_qkv<<<gGemm, blk, 0, stream>>>(k, Wk, xk, 1.0f);
  mha_proj_qkv<<<gGemm, blk, 0, stream>>>(v, Wv, xv, 1.0f);
  mha_flash_attn<<<gAttn, blk, 0, stream>>>(xq, xk, xv, xo);
  mha_proj_out<<<gGemm, blk, 0, stream>>>(xo, Wo, out);
}
